// DeepSpeedSelfAttention_49469433315618
// MI455X (gfx1250) — matrix operand order repaired and checked by reading, not compiled, not run on hardware
//
#include <hip/hip_runtime.h>

// ---------------------------------------------------------------- types
typedef __bf16 bf16_t;
typedef __attribute__((ext_vector_type(16))) __bf16 v16bf;
typedef __attribute__((ext_vector_type(8)))  __bf16 v8bf;
typedef __attribute__((ext_vector_type(8)))  float  v8f;
typedef __attribute__((ext_vector_type(4)))  float  v4f;
typedef int async_v4i __attribute__((vector_size(16)));   // matches builtin proto

#define AS1 __attribute__((address_space(1)))
#define AS3 __attribute__((address_space(3)))

#define H   4096
#define NH  32
#define HD  128
#define BB  2
#define SS  1024
#define ROWS (BB*SS)   // 2048
#define H3  (3*H)      // 12288

// fp32 -> bf16 round-to-nearest-even via bit ops
static __device__ __forceinline__ bf16_t f2bf(float f) {
    union { float f; unsigned u; } a; a.f = f;
    unsigned u = a.u + 0x7fffu + ((a.u >> 16) & 1u);
    union { unsigned short s; bf16_t b; } c; c.s = (unsigned short)(u >> 16);
    return c.b;
}

static __device__ __forceinline__ v16bf combine16(v8bf a, v8bf b) {
    return __builtin_shufflevector(a, b, 0,1,2,3,4,5,6,7,8,9,10,11,12,13,14,15);
}

// ---------------------------------------------------------------- async LDS staging
// gfx1250 async copy (ASYNCcnt engine). Guarded: falls back to synchronous
// load + ds_store if the builtins are absent in this toolchain.
#if __has_builtin(__builtin_amdgcn_global_load_async_to_lds_b128)
#define HAVE_ASYNC_LDS 1
static __device__ __forceinline__ void stage16(const bf16_t* g, bf16_t* l) {
    // AS(1) global keeps its 64-bit value; generic LDS pointer's low 32 bits
    // are the LDS byte offset (ISA 10.2).
    __builtin_amdgcn_global_load_async_to_lds_b128(
        (AS1 async_v4i*)(unsigned long long)(uintptr_t)g,
        (AS3 async_v4i*)(unsigned)(uintptr_t)l, 0, 0);
}
static __device__ __forceinline__ void stage_wait0() {
#if __has_builtin(__builtin_amdgcn_s_wait_asynccnt)
    __builtin_amdgcn_s_wait_asynccnt(0);
#else
    asm volatile("s_wait_asynccnt 0" ::: "memory");
#endif
}
static __device__ __forceinline__ void stage_wait1() {
#if __has_builtin(__builtin_amdgcn_s_wait_asynccnt)
    __builtin_amdgcn_s_wait_asynccnt(1);
#else
    asm volatile("s_wait_asynccnt 1" ::: "memory");
#endif
}
#else
static __device__ __forceinline__ void stage16(const bf16_t* g, bf16_t* l) {
    v8bf t = *(const v8bf*)g;
    *(v8bf*)l = t;
}
static __device__ __forceinline__ void stage_wait0() {}   // __syncthreads flushes DScnt
static __device__ __forceinline__ void stage_wait1() {}
#endif

// ---------------------------------------------------------------- WMMA helpers
// A-fragment, 16x32 bf16 (M x K) from global, row-major, leading dim ld.
// ISA 7.12.2: lane<16 : row=lane,    K = kb+{0..7}, kb+16+{0..7}
//             lane>=16: row=lane-16, K = kb+{8..15}, kb+24+{0..7}
static __device__ __forceinline__ v16bf load_a_frag(const bf16_t* __restrict__ base,
                                                    int ld, int m, int kb, int lane) {
    int half = lane >> 4;
    int row  = m + (lane & 15);
    const bf16_t* p = base + (size_t)row * ld + kb + half * 8;
    return combine16(*(const v8bf*)p, *(const v8bf*)(p + 16));
}

// A-fragment from a 32x32 LDS slab (row-major, 32 elements per row)
static __device__ __forceinline__ v16bf lds_a_frag(const bf16_t* sm, int mt, int lane) {
    int half = lane >> 4, l15 = lane & 15;
    const bf16_t* p = sm + (mt * 16 + l15) * 32 + half * 8;
    return combine16(*(const v8bf*)p, *(const v8bf*)(p + 16));
}

// B-fragment, 32x16 bf16 (K x N) given B^T stored [N][K] row-major:
// lane<16: col n=lane holds K=kb+{0..7},kb+16+{0..7} ; lane>=16: col n=lane-16 holds K=kb+{8..15},kb+24+{0..7}
static __device__ __forceinline__ v16bf load_b_frag(const bf16_t* __restrict__ baseT,
                                                    int ld, int n, int kb, int lane) {
    int half = lane >> 4;
    int row  = n + (lane & 15);
    const bf16_t* p = baseT + (size_t)row * ld + kb + half * 8;
    return combine16(*(const v8bf*)p, *(const v8bf*)(p + 16));
}

static __device__ __forceinline__ v8f wmma_bf16(v16bf a, v16bf b, v8f c) {
    return __builtin_amdgcn_wmma_f32_16x16x32_bf16(false, a, false, b,
                                                   (short)0, c, false, false);
}

// ---------------------------------------------------------------- 1) LayerNorm -> bf16
__global__ __launch_bounds__(256) void ln_bf16_kernel(const float* __restrict__ x,
                                                      const float* __restrict__ w,
                                                      const float* __restrict__ b,
                                                      bf16_t* __restrict__ out) {
    __shared__ float red[16];
    int row = blockIdx.x;
    const float* xr = x + (size_t)row * H;
    int base = threadIdx.x * 16;
    float v[16];
#pragma unroll
    for (int i = 0; i < 4; ++i) {
        v4f t = *(const v4f*)(xr + base + i * 4);
#pragma unroll
        for (int j = 0; j < 4; ++j) v[i * 4 + j] = t[j];
    }
    float s = 0.f, q = 0.f;
#pragma unroll
    for (int i = 0; i < 16; ++i) { s += v[i]; q += v[i] * v[i]; }
#pragma unroll
    for (int off = 1; off < 32; off <<= 1) {
        s += __shfl_xor(s, off, 32);
        q += __shfl_xor(q, off, 32);
    }
    int wid = threadIdx.x >> 5, lane = threadIdx.x & 31;
    if (lane == 0) { red[wid] = s; red[8 + wid] = q; }
    __syncthreads();
    if (threadIdx.x == 0) {
        float ts = 0.f, tq = 0.f;
#pragma unroll
        for (int i = 0; i < 8; ++i) { ts += red[i]; tq += red[8 + i]; }
        red[0] = ts; red[1] = tq;
    }
    __syncthreads();
    float mu  = red[0] * (1.0f / H);
    float var = red[1] * (1.0f / H) - mu * mu;
    float rs  = rsqrtf(var + 1e-5f);
    bf16_t* orow = out + (size_t)row * H;
#pragma unroll
    for (int i = 0; i < 16; ++i) {
        int c = base + i;
        orow[c] = f2bf((v[i] - mu) * rs * w[c] + b[c]);
    }
}

// ---------------------------------------------------------------- 2) W[K][N] -> Wt[N][K] bf16
__global__ __launch_bounds__(256) void transpose_bf16_kernel(const float* __restrict__ W,
                                                             bf16_t* __restrict__ Wt,
                                                             int K, int N) {
    __shared__ float tile[32][33];
    int n0 = blockIdx.x * 32, k0 = blockIdx.y * 32;
    int tx = threadIdx.x & 31, ty = threadIdx.x >> 5;
#pragma unroll
    for (int i = 0; i < 4; ++i)
        tile[ty + i * 8][tx] = W[(size_t)(k0 + ty + i * 8) * N + n0 + tx];
    __syncthreads();
#pragma unroll
    for (int i = 0; i < 4; ++i)
        Wt[(size_t)(n0 + ty + i * 8) * K + k0 + tx] = f2bf(tile[tx][ty + i * 8]);
}

// ---------------------------------------------------------------- 3) QKV GEMM
// C/D layout: element r of v8f in lane L -> row = r + 8*(L>=16), col = L&15
static __device__ __forceinline__ void store_qkv(v8f acc, int m0, int n0, int lane,
                                                 const float* __restrict__ bias,
                                                 bf16_t* __restrict__ qb,
                                                 bf16_t* __restrict__ kb,
                                                 bf16_t* __restrict__ vtb) {
    int half = lane >> 4, col = lane & 15;
    int n = n0 + col;
    float bv = bias[n];
    int which = n >> 12;
    int c = n & (H - 1);
    int head = c >> 7, hd = c & 127;
    const float NF = 0.08838834764831845f;  // 1/sqrt(128)
#pragma unroll
    for (int r = 0; r < 8; ++r) {
        int gm = m0 + r + 8 * half;
        int b_ = gm >> 10, s = gm & (SS - 1);
        float val = acc[r] + bv;
        if (which == 0)
            qb[(((size_t)(b_ * NH + head)) * SS + s) * HD + hd] = f2bf(val * NF);
        else if (which == 1)
            kb[(((size_t)(b_ * NH + head)) * SS + s) * HD + hd] = f2bf(val);
        else
            vtb[(((size_t)(b_ * NH + head)) * HD + hd) * SS + s] = f2bf(val);
    }
}

// Block = 4 waves, shared 32-row A slab staged in LDS (double buffered,
// async-copy engine); each wave owns a distinct 32-col strip of Wt.
__global__ __launch_bounds__(128) void qkv_gemm_kernel(const bf16_t* __restrict__ A,   // [2048][4096]
                                                       const bf16_t* __restrict__ Wt,  // [12288][4096]
                                                       const float*  __restrict__ bias,// [12288]
                                                       bf16_t* __restrict__ qb,
                                                       bf16_t* __restrict__ kb,
                                                       bf16_t* __restrict__ vtb) {
    __shared__ alignas(128) bf16_t asl[2][32 * 32];
    int tid  = threadIdx.x;
    int lane = tid & 31, wid = tid >> 5;
    int mg   = blockIdx.x / 96;
    int n0   = ((blockIdx.x % 96) * 4 + wid) * 32;
    int m0   = mg * 32;

    int srow = tid >> 2, schunk = tid & 3;           // 32 rows x 4 chunks of 16B
    const bf16_t* gsrc = A + (size_t)(m0 + srow) * H + schunk * 8;
    bf16_t* lds0 = &asl[0][srow * 32 + schunk * 8];
    bf16_t* lds1 = &asl[1][srow * 32 + schunk * 8];

    v8f a00 = {}, a01 = {}, a10 = {}, a11 = {};
    const int NSTEP = H / 32;                        // 128
    stage16(gsrc, lds0);                             // prologue: k-step 0
    for (int i = 0; i < NSTEP; ++i) {
        int kbs = i * 32;
        if (i + 1 < NSTEP) {
            stage16(gsrc + kbs + 32, ((i + 1) & 1) ? lds1 : lds0);
            stage_wait1();                           // own step-i portion done
        } else {
            stage_wait0();
        }
        __syncthreads();                             // all portions visible
        const bf16_t* sm = asl[i & 1];
        __builtin_prefetch(Wt + (size_t)(n0 + (lane & 15)) * H + kbs + 128, 0, 1);
        v16bf af0 = lds_a_frag(sm, 0, lane);
        v16bf af1 = lds_a_frag(sm, 1, lane);
        v16bf bf0 = load_b_frag(Wt, H, n0,      kbs, lane);
        v16bf bf1 = load_b_frag(Wt, H, n0 + 16, kbs, lane);
        a00 = wmma_bf16(af0, bf0, a00);
        a01 = wmma_bf16(af0, bf1, a01);
        a10 = wmma_bf16(af1, bf0, a10);
        a11 = wmma_bf16(af1, bf1, a11);
        __syncthreads();                             // release buffer i&1
    }
    store_qkv(a00, m0,      n0,      lane, bias, qb, kb, vtb);
    store_qkv(a01, m0,      n0 + 16, lane, bias, qb, kb, vtb);
    store_qkv(a10, m0 + 16, n0,      lane, bias, qb, kb, vtb);
    store_qkv(a11, m0 + 16, n0 + 16, lane, bias, qb, kb, vtb);
}

// ---------------------------------------------------------------- 4) causal flash attention
__global__ __launch_bounds__(32) void attn_kernel(const bf16_t* __restrict__ qb,  // [b,h,s,hd] pre-scaled
                                                  const bf16_t* __restrict__ kb,  // [b,h,s,hd]
                                                  const bf16_t* __restrict__ vtb, // [b,h,hd,s]
                                                  const float*  __restrict__ mask,// [B,1,1,S]
                                                  bf16_t* __restrict__ ctx) {     // [b,s,H]
    __shared__ alignas(64) bf16_t pls[16 * 32];
    int lane = threadIdx.x;
    int qt   = blockIdx.x & 63;
    int head = (blockIdx.x >> 6) & 31;
    int b_   = blockIdx.x >> 11;
    int q0   = qt * 16;
    int half = lane >> 4, l15 = lane & 15;

    const bf16_t* Q  = qb  + ((size_t)(b_ * NH + head)) * SS * HD;
    const bf16_t* K  = kb  + ((size_t)(b_ * NH + head)) * SS * HD;
    const bf16_t* VT = vtb + ((size_t)(b_ * NH + head)) * HD * SS;

    v16bf qf[4];
#pragma unroll
    for (int hc = 0; hc < 4; ++hc) qf[hc] = load_a_frag(Q, HD, q0, hc * 32, lane);

    v8f ctxa[8];
    v8f zero = {};
#pragma unroll
    for (int ht = 0; ht < 8; ++ht) ctxa[ht] = zero;
    float mrun[8], lrun[8];
#pragma unroll
    for (int r = 0; r < 8; ++r) { mrun[r] = -1e30f; lrun[r] = 0.f; }

    int q_last = q0 + 15;
    for (int kv0 = 0; kv0 <= q_last; kv0 += 32) {
        v8f s0 = zero, s1 = zero;
#pragma unroll
        for (int hc = 0; hc < 4; ++hc) {
            v16bf b0 = load_b_frag(K, HD, kv0,      hc * 32, lane);
            v16bf b1 = load_b_frag(K, HD, kv0 + 16, hc * 32, lane);
            s0 = wmma_bf16(qf[hc], b0, s0);
            s1 = wmma_bf16(qf[hc], b1, s1);
        }
        int c0 = kv0 + l15, c1 = kv0 + 16 + l15;
        float mk0 = mask[b_ * SS + c0];
        float mk1 = mask[b_ * SS + c1];
#pragma unroll
        for (int r = 0; r < 8; ++r) {
            int row = q0 + r + 8 * half;
            s0[r] += mk0 + ((c0 > row) ? -10000.0f : 0.0f);
            s1[r] += mk1 + ((c1 > row) ? -10000.0f : 0.0f);
        }
#pragma unroll
        for (int r = 0; r < 8; ++r) {
            float mx = fmaxf(s0[r], s1[r]);
#pragma unroll
            for (int off = 1; off < 16; off <<= 1) mx = fmaxf(mx, __shfl_xor(mx, off, 32));
            float mnew  = fmaxf(mrun[r], mx);
            float alpha = __expf(mrun[r] - mnew);
            float p0 = __expf(s0[r] - mnew);
            float p1 = __expf(s1[r] - mnew);
            float ps = p0 + p1;
#pragma unroll
            for (int off = 1; off < 16; off <<= 1) ps += __shfl_xor(ps, off, 32);
            lrun[r] = lrun[r] * alpha + ps;
            mrun[r] = mnew;
            s0[r] = p0; s1[r] = p1;
#pragma unroll
            for (int ht = 0; ht < 8; ++ht) ctxa[ht][r] *= alpha;
        }
#pragma unroll
        for (int r = 0; r < 8; ++r) {
            int row = r + 8 * half;
            pls[row * 32 + l15]      = f2bf(s0[r]);
            pls[row * 32 + 16 + l15] = f2bf(s1[r]);
        }
        __syncthreads();
        const bf16_t* pr = pls + l15 * 32 + half * 8;
        v16bf pf = combine16(*(const v8bf*)pr, *(const v8bf*)(pr + 16));
        __syncthreads();
#pragma unroll
        for (int ht = 0; ht < 8; ++ht) {
            v16bf bv = load_b_frag(VT, SS, ht * 16, kv0, lane);
            ctxa[ht] = wmma_bf16(pf, bv, ctxa[ht]);
        }
    }
    bf16_t* cb = ctx + (size_t)b_ * SS * H;
#pragma unroll
    for (int ht = 0; ht < 8; ++ht) {
#pragma unroll
        for (int r = 0; r < 8; ++r) {
            int srow = q0 + r + 8 * half;
            float inv = 1.0f / lrun[r];
            cb[(size_t)srow * H + head * HD + ht * 16 + l15] = f2bf(ctxa[ht][r] * inv);
        }
    }
}

// ---------------------------------------------------------------- 5) output projection
__global__ __launch_bounds__(128) void oproj_kernel(const bf16_t* __restrict__ A,   // ctx [2048][4096]
                                                    const bf16_t* __restrict__ Wt,  // [4096][4096] (N,K)
                                                    float* __restrict__ out) {      // [2048][4096]
    __shared__ alignas(128) bf16_t asl[2][32 * 32];
    int tid  = threadIdx.x;
    int lane = tid & 31, wid = tid >> 5;
    int mg   = blockIdx.x / 32;
    int n0   = ((blockIdx.x % 32) * 4 + wid) * 32;
    int m0   = mg * 32;

    int srow = tid >> 2, schunk = tid & 3;
    const bf16_t* gsrc = A + (size_t)(m0 + srow) * H + schunk * 8;
    bf16_t* lds0 = &asl[0][srow * 32 + schunk * 8];
    bf16_t* lds1 = &asl[1][srow * 32 + schunk * 8];

    v8f a00 = {}, a01 = {}, a10 = {}, a11 = {};
    const int NSTEP = H / 32;
    stage16(gsrc, lds0);
    for (int i = 0; i < NSTEP; ++i) {
        int kbs = i * 32;
        if (i + 1 < NSTEP) {
            stage16(gsrc + kbs + 32, ((i + 1) & 1) ? lds1 : lds0);
            stage_wait1();
        } else {
            stage_wait0();
        }
        __syncthreads();
        const bf16_t* sm = asl[i & 1];
        __builtin_prefetch(Wt + (size_t)(n0 + (lane & 15)) * H + kbs + 128, 0, 1);
        v16bf af0 = lds_a_frag(sm, 0, lane);
        v16bf af1 = lds_a_frag(sm, 1, lane);
        v16bf bf0 = load_b_frag(Wt, H, n0,      kbs, lane);
        v16bf bf1 = load_b_frag(Wt, H, n0 + 16, kbs, lane);
        a00 = wmma_bf16(af0, bf0, a00);
        a01 = wmma_bf16(af0, bf1, a01);
        a10 = wmma_bf16(af1, bf0, a10);
        a11 = wmma_bf16(af1, bf1, a11);
        __syncthreads();
    }
    int half = lane >> 4, col = lane & 15;
#pragma unroll
    for (int r = 0; r < 8; ++r) {
        int gm = m0 + r + 8 * half;
        out[(size_t)gm * H + n0 + col]      = a00[r];
        out[(size_t)gm * H + n0 + 16 + col] = a01[r];
        out[(size_t)(gm + 16) * H + n0 + col]      = a10[r];
        out[(size_t)(gm + 16) * H + n0 + 16 + col] = a11[r];
    }
}

// ---------------------------------------------------------------- launch
extern "C" void kernel_launch(void* const* d_in, const int* in_sizes, int n_in,
                              void* d_out, int out_size, void* d_ws, size_t ws_size,
                              hipStream_t stream) {
    (void)in_sizes; (void)n_in; (void)out_size; (void)ws_size;
    const float* x       = (const float*)d_in[0];
    const float* mask    = (const float*)d_in[1];
    const float* norm_w  = (const float*)d_in[2];
    const float* norm_b  = (const float*)d_in[3];
    const float* qkv_w   = (const float*)d_in[4];
    const float* qkv_b   = (const float*)d_in[5];
    const float* attn_ow = (const float*)d_in[6];

    char* ws = (char*)d_ws;
    bf16_t* ln  = (bf16_t*)(ws);                          // 16 MiB  [2048][4096]
    bf16_t* qwt = (bf16_t*)(ws + (16ull  << 20));         // 96 MiB  [12288][4096]
    bf16_t* owt = (bf16_t*)(ws + (112ull << 20));         // 32 MiB  [4096][4096]
    bf16_t* qb  = (bf16_t*)(ws + (144ull << 20));         // 16 MiB  [b,h,s,hd]
    bf16_t* kb  = (bf16_t*)(ws + (160ull << 20));         // 16 MiB  [b,h,s,hd]
    bf16_t* vtb = (bf16_t*)(ws + (176ull << 20));         // 16 MiB  [b,h,hd,s]
    bf16_t* ctx = (bf16_t*)(ws + (192ull << 20));         // 16 MiB  [b,s,H]

    ln_bf16_kernel<<<ROWS, 256, 0, stream>>>(x, norm_w, norm_b, ln);
    transpose_bf16_kernel<<<dim3(H3 / 32, H / 32), 256, 0, stream>>>(qkv_w, qwt, H, H3);
    transpose_bf16_kernel<<<dim3(H / 32, H / 32), 256, 0, stream>>>(attn_ow, owt, H, H);
    qkv_gemm_kernel<<<64 * 96, 128, 0, stream>>>(ln, qwt, qkv_b, qb, kb, vtb);
    attn_kernel<<<BB * NH * (SS / 16), 32, 0, stream>>>(qb, kb, vtb, mask, ctx);
    oproj_kernel<<<64 * 32, 128, 0, stream>>>(ctx, owt, (float*)d_out);
}
